// SelfAttention_34462817583504
// MI455X (gfx1250) — matrix operand order repaired and checked by reading, not compiled, not run on hardware
//
#include <hip/hip_runtime.h>

typedef __attribute__((ext_vector_type(16))) _Float16 v16h;
typedef __attribute__((ext_vector_type(8)))  _Float16 v8h;
typedef __attribute__((ext_vector_type(8)))  float    v8f;

#define WM_B  4
#define WM_H  8
#define WM_S  2048
#define WM_HD 64
#define WM_D  512   // = H*HD

// 16-lane (DPP row) butterfly reduction: xor1, xor2, half_mirror, mirror.
// Keeps the two halves of a wave independent, matching the WMMA C layout
// (row = r + 8*half, 16 lanes per row).
#define DPP_F(x, ctrl) \
    __int_as_float(__builtin_amdgcn_update_dpp(0, __float_as_int(x), ctrl, 0xf, 0xf, false))

__device__ __forceinline__ float rowmax16(float x) {
    x = fmaxf(x, DPP_F(x, 0xB1));   // quad_perm [1,0,3,2]  (xor 1)
    x = fmaxf(x, DPP_F(x, 0x4E));   // quad_perm [2,3,0,1]  (xor 2)
    x = fmaxf(x, DPP_F(x, 0x141));  // row_half_mirror      (combine quads)
    x = fmaxf(x, DPP_F(x, 0x140));  // row_mirror           (combine octs)
    return x;
}
__device__ __forceinline__ float rowsum16(float x) {
    x += DPP_F(x, 0xB1);
    x += DPP_F(x, 0x4E);
    x += DPP_F(x, 0x141);
    x += DPP_F(x, 0x140);
    return x;
}

#define SHUF16(a, b) __builtin_shufflevector(a, b, 0,1,2,3,4,5,6,7,8,9,10,11,12,13,14,15)

// ---------------------------------------------------------------------------
// Prep: f32 -> f16 with scale (used for q (x0.125), k, and W).
// ---------------------------------------------------------------------------
__global__ __launch_bounds__(256) void cvt_scale_kernel(const float* __restrict__ src,
                                                        _Float16* __restrict__ dst,
                                                        float scale)
{
    size_t i = ((size_t)blockIdx.x * 256 + threadIdx.x) * 8;
    #pragma unroll
    for (int j = 0; j < 8; ++j) dst[i + j] = (_Float16)(src[i + j] * scale);
}

// ---------------------------------------------------------------------------
// Prep: per-(b,h) transpose of V to vt[bh][d][s] as f16 (LDS-tiled 64x64).
// ---------------------------------------------------------------------------
__global__ __launch_bounds__(256) void vtrans_kernel(const float* __restrict__ v,
                                                     _Float16* __restrict__ vt)
{
    __shared__ _Float16 t[64][72];
    const int tid = threadIdx.x;
    const int bh  = blockIdx.y;
    const int s0  = blockIdx.x * 64;
    const size_t base = (size_t)bh * WM_S * WM_HD;

    const int row = tid >> 2;          // [0,64)
    const int col = (tid & 3) * 16;    // runs of 16

    const float* p = v + base + (size_t)(s0 + row) * WM_HD + col;
    #pragma unroll
    for (int i = 0; i < 16; ++i) t[row][col + i] = (_Float16)p[i];
    __syncthreads();

    _Float16* q = vt + base + (size_t)row * WM_S + s0 + col;   // row = d here
    #pragma unroll
    for (int i = 0; i < 16; ++i) q[i] = t[col + i][row];
}

// ---------------------------------------------------------------------------
// Kernel 1: flash attention. One (b,h) per blockIdx.y; 128 query rows per
// block (8 waves x 16 rows); 64-key tiles staged through LDS in f16.
// PRE=true: q16/k16/vt16 are pre-converted f16 (q pre-scaled), staging is
// pure b128 copies. PRE=false: convert/transpose in-loop from f32 (fallback
// when workspace is too small).
// Output: f16 X in the reference's raw-reshape layout:
// X[b*2048 + h*256 + s/8][(s%8)*64 + d].
// ---------------------------------------------------------------------------
template <bool PRE>
__global__ __launch_bounds__(256) void attn_kernel(
    const float* __restrict__ qg, const float* __restrict__ kg,
    const float* __restrict__ vg,
    const _Float16* __restrict__ q16, const _Float16* __restrict__ k16,
    const _Float16* __restrict__ vt16, _Float16* __restrict__ X)
{
    __shared__ _Float16 Kt[64][72];     // [key][d]
    __shared__ _Float16 VtT[64][72];    // [d][key]
    __shared__ _Float16 Pst[8][16][72]; // per-wave P restage, [m][key_local]

    const int tid  = threadIdx.x;
    const int lane = tid & 31;
    const int wave = tid >> 5;
    const int hg   = lane >> 4;
    const int n16  = lane & 15;

    const int bh = blockIdx.y;
    const int b  = bh >> 3;             // H = 8
    const int h  = bh & 7;
    const size_t base = (size_t)bh * WM_S * WM_HD;
    const int m0 = blockIdx.x * 128 + wave * 16;

    // ---- Q fragments (A layout: lane m = lane&15; e -> k = (e>>3)*16+8*hg+(e&7))
    v16h aq[2];
    if constexpr (PRE) {
        const _Float16* qrow = q16 + base + (size_t)(m0 + n16) * WM_HD;
        #pragma unroll
        for (int f = 0; f < 2; ++f) {
            v8h a0 = *(const v8h*)(qrow + 32 * f + 8 * hg);
            v8h a1 = *(const v8h*)(qrow + 32 * f + 8 * hg + 16);
            aq[f] = SHUF16(a0, a1);
        }
    } else {
        const float* qrow = qg + base + (size_t)(m0 + n16) * WM_HD;
        #pragma unroll
        for (int f = 0; f < 2; ++f) {
            const float* qp = qrow + 32 * f + 8 * hg;
            #pragma unroll
            for (int e = 0; e < 8; ++e) {
                aq[f][e]     = (_Float16)(qp[e]      * 0.125f);
                aq[f][e + 8] = (_Float16)(qp[16 + e] * 0.125f);
            }
        }
    }

    float mrun[8], lrun[8];
    v8f oacc[4];
    #pragma unroll
    for (int r = 0; r < 8; ++r) { mrun[r] = -1e30f; lrun[r] = 0.f; }
    #pragma unroll
    for (int t = 0; t < 4; ++t) { v8f z = {}; oacc[t] = z; }

    const int row = tid >> 2;           // staging: [0,64)
    const int col = (tid & 3) * 16;     // staging: runs of 16

    for (int j0 = 0; j0 < WM_S; j0 += 64) {
        __syncthreads();
        // ---- stage 64-key K tile [key][d] and V^T tile [d][key] ----
        if constexpr (PRE) {
            const _Float16* kp = k16  + base + (size_t)(j0 + row) * WM_HD + col;
            const _Float16* vp = vt16 + base + (size_t)row * WM_S + j0 + col;
            #pragma unroll
            for (int i = 0; i < 16; i += 8) {
                *(v8h*)&Kt[row][col + i]  = *(const v8h*)(kp + i);
                *(v8h*)&VtT[row][col + i] = *(const v8h*)(vp + i);
            }
        } else {
            const float* kp = kg + base + (size_t)(j0 + row) * WM_HD + col;
            const float* vp = vg + base + (size_t)(j0 + row) * WM_HD + col;
            #pragma unroll
            for (int i = 0; i < 16; ++i) {
                Kt[row][col + i]  = (_Float16)kp[i];
                VtT[col + i][row] = (_Float16)vp[i];
            }
        }
        __syncthreads();

        // ---- scores: four 16x16 tiles covering 64 keys; K-dim = HD = 64 ----
        v8f c[4];
        #pragma unroll
        for (int sc = 0; sc < 4; ++sc) {
            v8f acc = {};
            #pragma unroll
            for (int f = 0; f < 2; ++f) {
                // B layout: lane n = lane&15; element e -> k = (e>>3)*16 + 8*half + (e&7)
                v8h b0 = *(const v8h*)&Kt[16 * sc + n16][32 * f + 8 * hg];
                v8h b1 = *(const v8h*)&Kt[16 * sc + n16][32 * f + 8 * hg + 16];
                v16h bk = SHUF16(b0, b1);
                acc = __builtin_amdgcn_wmma_f32_16x16x32_f16(false, aq[f], false, bk,
                                                             (short)0, acc, false, false);
            }
            c[sc] = acc;
        }

        // ---- online softmax, DPP reductions (no DS traffic) ----
        #pragma unroll
        for (int r = 0; r < 8; ++r) {
            float mx = fmaxf(fmaxf(c[0][r], c[1][r]), fmaxf(c[2][r], c[3][r]));
            mx = rowmax16(mx);
            float mnew = fmaxf(mrun[r], mx);
            float al   = __expf(mrun[r] - mnew);
            mrun[r] = mnew;
            float rs = 0.f;
            #pragma unroll
            for (int sc = 0; sc < 4; ++sc) {
                float p = __expf(c[sc][r] - mnew);
                c[sc][r] = p;
                rs += p;
            }
            rs = rowsum16(rs);
            lrun[r] = lrun[r] * al + rs;
            oacc[0][r] *= al; oacc[1][r] *= al; oacc[2][r] *= al; oacc[3][r] *= al;
        }

        // ---- restage P: C layout -> LDS -> A layout ----
        #pragma unroll
        for (int r = 0; r < 8; ++r) {
            int m = r + (hg << 3);
            #pragma unroll
            for (int sc = 0; sc < 4; ++sc)
                Pst[wave][m][16 * sc + n16] = (_Float16)c[sc][r];
        }
        __syncthreads();
        v16h ap[2];
        #pragma unroll
        for (int f = 0; f < 2; ++f) {
            v8h p0 = *(const v8h*)&Pst[wave][n16][32 * f + 8 * hg];
            v8h p1 = *(const v8h*)&Pst[wave][n16][32 * f + 16 + 8 * hg];
            ap[f] = SHUF16(p0, p1);
        }

        // ---- O += P * V  (N = 64 -> four 16-col tiles, K = 64 keys) ----
        #pragma unroll
        for (int t = 0; t < 4; ++t) {
            #pragma unroll
            for (int f = 0; f < 2; ++f) {
                v8h v0 = *(const v8h*)&VtT[16 * t + n16][32 * f + 8 * hg];
                v8h v1 = *(const v8h*)&VtT[16 * t + n16][32 * f + 8 * hg + 16];
                v16h vb = SHUF16(v0, v1);
                oacc[t] = __builtin_amdgcn_wmma_f32_16x16x32_f16(false, ap[f], false, vb,
                                                                 (short)0, oacc[t], false, false);
            }
        }
    }

    // ---- normalize and store into raw-reshape layout X[8192][512] (f16) ----
    #pragma unroll
    for (int r = 0; r < 8; ++r) {
        int s = m0 + r + (hg << 3);
        float inv = 1.0f / lrun[r];
        size_t prow = (size_t)b * 2048 + (size_t)h * 256 + (s >> 3);
        int colb = (s & 7) * 64;
        #pragma unroll
        for (int t = 0; t < 4; ++t)
            X[prow * WM_D + colb + 16 * t + n16] = (_Float16)(oacc[t][r] * inv);
    }
}

// ---------------------------------------------------------------------------
// Kernel 2: Y[8192,512] = X[8192,512] * W^T + bias.
// One 16x16 output tile per wave; K = 512 in 16 WMMA steps, all-f16 operands.
// ---------------------------------------------------------------------------
__global__ __launch_bounds__(256) void proj_kernel(
    const _Float16* __restrict__ X, const _Float16* __restrict__ W16,
    const float* __restrict__ bias, float* __restrict__ out)
{
    const int lane = threadIdx.x & 31;
    const int wave = threadIdx.x >> 5;
    const int hg   = lane >> 4;
    const int n16  = lane & 15;

    int tile = blockIdx.x * 8 + wave;   // 2048 blocks * 8 waves = 16384 tiles
    int pi   = tile >> 5;               // 512 row tiles
    int ji   = tile & 31;               // 32 col tiles

    const _Float16* xrow = X   + (size_t)(pi * 16 + n16) * WM_D;
    const _Float16* wrow = W16 + (size_t)(ji * 16 + n16) * WM_D;

    v8f acc = {};
    for (int cc = 0; cc < WM_D; cc += 32) {
        v8h x0 = *(const v8h*)(xrow + cc + 8 * hg);
        v8h x1 = *(const v8h*)(xrow + cc + 16 + 8 * hg);
        v16h ax = SHUF16(x0, x1);
        v8h w0 = *(const v8h*)(wrow + cc + 8 * hg);
        v8h w1 = *(const v8h*)(wrow + cc + 8 * hg + 16);
        v16h bw = SHUF16(w0, w1);
        acc = __builtin_amdgcn_wmma_f32_16x16x32_f16(false, ax, false, bw,
                                                     (short)0, acc, false, false);
    }
    float bv = bias[ji * 16 + n16];
    #pragma unroll
    for (int r = 0; r < 8; ++r)
        out[(size_t)(pi * 16 + r + (hg << 3)) * WM_D + ji * 16 + n16] = acc[r] + bv;
}

extern "C" void kernel_launch(void* const* d_in, const int* in_sizes, int n_in,
                              void* d_out, int out_size, void* d_ws, size_t ws_size,
                              hipStream_t stream) {
    const float* q    = (const float*)d_in[0];
    const float* k    = (const float*)d_in[1];
    const float* v    = (const float*)d_in[2];
    const float* W    = (const float*)d_in[3];
    const float* bias = (const float*)d_in[4];
    float* out = (float*)d_out;

    const size_t NX = (size_t)WM_B * WM_S * WM_D;        // 4,194,304
    const size_t NW = (size_t)WM_D * WM_D;               //   262,144
    const size_t NQ = (size_t)WM_B * WM_H * WM_S * WM_HD;// 4,194,304

    _Float16* X    = (_Float16*)d_ws;        // 8 MB
    _Float16* W16  = X + NX;                 // 512 KB
    _Float16* q16  = W16 + NW;               // 8 MB (optional)
    _Float16* k16  = q16 + NQ;               // 8 MB (optional)
    _Float16* vt16 = k16 + NQ;               // 8 MB (optional)

    const bool pre = ws_size >= (NX + NW + 3 * NQ) * sizeof(_Float16);

    cvt_scale_kernel<<<(unsigned)(NW / (256 * 8)), 256, 0, stream>>>(W, W16, 1.0f);

    dim3 g1(WM_S / 128, WM_B * WM_H);                    // (16, 32)
    if (pre) {
        cvt_scale_kernel<<<(unsigned)(NQ / (256 * 8)), 256, 0, stream>>>(q, q16, 0.125f);
        cvt_scale_kernel<<<(unsigned)(NQ / (256 * 8)), 256, 0, stream>>>(k, k16, 1.0f);
        dim3 gt(WM_S / 64, WM_B * WM_H);                 // (32, 32)
        vtrans_kernel<<<gt, 256, 0, stream>>>(v, vt16);
        attn_kernel<true><<<g1, 256, 0, stream>>>(q, k, v, q16, k16, vt16, X);
    } else {
        attn_kernel<false><<<g1, 256, 0, stream>>>(q, k, v, q16, k16, vt16, X);
    }
    proj_kernel<<<2048, 256, 0, stream>>>(X, W16, bias, out);
}
